// Layer_15736760173151
// MI455X (gfx1250) — matrix operand order repaired and checked by reading, not compiled, not run on hardware
//
#include <hip/hip_runtime.h>
#include <math.h>

#define NFEAT 512
#define NBATCH 4096

typedef __attribute__((ext_vector_type(16))) __bf16 v16bf;
typedef __attribute__((ext_vector_type(8)))  float  v8f;

union FragBF { unsigned int u[8]; v16bf v; };

// ---- Cl(3,0) Cayley table (a-major), verified against blade algebra ----
// blades: 0:1  1:e1  2:e2  3:e3  4:e12  5:e13  6:e23  7:e123
constexpr int TA[64] = {
  0,0,0,0,0,0,0,0, 1,1,1,1,1,1,1,1, 2,2,2,2,2,2,2,2, 3,3,3,3,3,3,3,3,
  4,4,4,4,4,4,4,4, 5,5,5,5,5,5,5,5, 6,6,6,6,6,6,6,6, 7,7,7,7,7,7,7,7 };
constexpr int TB[64] = {
  0,1,2,3,4,5,6,7, 0,1,2,3,4,5,6,7, 0,1,2,3,4,5,6,7, 0,1,2,3,4,5,6,7,
  0,1,2,3,4,5,6,7, 0,1,2,3,4,5,6,7, 0,1,2,3,4,5,6,7, 0,1,2,3,4,5,6,7 };
constexpr int TK[64] = {
  0,1,2,3,4,5,6,7,  1,0,4,5,2,3,7,6,  2,4,0,6,1,7,3,5,  3,5,6,0,7,1,2,4,
  4,2,1,7,0,6,5,3,  5,3,7,1,6,0,4,2,  6,7,3,2,5,4,0,1,  7,6,5,4,3,2,1,0 };
constexpr float TS[64] = {
  1,1,1,1,1,1,1,1,   1,1,1,1,1,1,1,1,   1,-1,1,1,-1,-1,1,-1,  1,-1,-1,1,1,-1,-1,1,
  1,-1,1,1,-1,-1,1,-1, 1,-1,-1,1,1,-1,-1,1, 1,1,-1,1,-1,1,-1,-1, 1,1,-1,1,-1,1,-1,-1 };
// path index into the 20 sorted (grade_a, grade_b, grade_out) triples
constexpr int TP[64] = {
  0,1,1,1,2,2,2,3,      4,5,6,6,7,7,8,9,      4,6,5,6,7,8,7,9,      4,6,6,5,8,7,7,9,
  10,11,11,12,13,14,14,15, 10,11,12,11,14,13,14,15, 10,12,11,11,14,14,13,15, 16,17,17,17,18,18,18,19 };
constexpr int GRD[8] = {0,1,1,1,2,2,2,3};

__device__ __forceinline__ unsigned short f32_to_bf16(float f) {
  unsigned int u = __builtin_bit_cast(unsigned int, f);
  u += 0x7FFFu + ((u >> 16) & 1u);       // round-to-nearest-even
  return (unsigned short)(u >> 16);
}

// ---------------------------------------------------------------------------
// Kernel 1: y[blade] = x[blade] @ W[grade(blade)] + bias   (bf16 WMMA, f32 acc)
// block: 128 threads (4 waves), block tile 128x64, wave tile 32x64, K-step 32
// double-buffered LDS; next tile staged in registers while WMMAs run
// ---------------------------------------------------------------------------
__global__ __launch_bounds__(128)
void ga_gemm_kernel(const float* __restrict__ x,
                    const float* __restrict__ lw,
                    const float* __restrict__ bias,
                    float* __restrict__ y) {
  __shared__ unsigned short As[2][128 * 32];   // [buf][row][k]
  __shared__ unsigned short Bs[2][64 * 32];    // [buf][col][k]  (W tile transposed)

  const int tid  = threadIdx.x;
  const int wave = tid >> 5;
  const int lane = tid & 31;
  const int m    = lane & 15;
  const int half = lane >> 4;

  const int blade   = blockIdx.z;
  const int grade   = (blade == 0) ? 0 : ((blade < 4) ? 1 : ((blade < 7) ? 2 : 3));
  const int rowBase = blockIdx.y * 128;
  const int colBase = blockIdx.x * 64;

  const float* xa = x  + (size_t)blade * NBATCH * NFEAT;
  const float* W  = lw + (size_t)grade * NFEAT * NFEAT;
  float*       ya = y  + (size_t)blade * NBATCH * NFEAT;

  v8f acc[2][4];
#pragma unroll
  for (int p = 0; p < 2; ++p)
#pragma unroll
    for (int j = 0; j < 4; ++j)
#pragma unroll
      for (int r = 0; r < 8; ++r) acc[p][j][r] = 0.0f;

  float4 ar[8];      // A stage: thread owns one full 32-wide row
  float  br[16];     // B stage

  // ---- prologue: load + convert + store K-tile 0 into buffer 0 ----
  {
    const float* srcA = xa + (size_t)(rowBase + tid) * NFEAT;
#pragma unroll
    for (int i = 0; i < 8; ++i) ar[i] = *(const float4*)(srcA + 4 * i);
#pragma unroll
    for (int i = 0; i < 16; ++i) {
      const int idx = tid + 128 * i;
      br[i] = W[(size_t)(idx >> 6) * NFEAT + colBase + (idx & 63)];
    }
    unsigned short* dA = As[0] + tid * 32;
#pragma unroll
    for (int i = 0; i < 8; ++i) {
      dA[4 * i + 0] = f32_to_bf16(ar[i].x);
      dA[4 * i + 1] = f32_to_bf16(ar[i].y);
      dA[4 * i + 2] = f32_to_bf16(ar[i].z);
      dA[4 * i + 3] = f32_to_bf16(ar[i].w);
    }
#pragma unroll
    for (int i = 0; i < 16; ++i) {
      const int idx = tid + 128 * i;
      Bs[0][(idx & 63) * 32 + (idx >> 6)] = f32_to_bf16(br[i]);
    }
  }
  __syncthreads();

  int buf = 0;
  for (int kk = 0; kk < NFEAT; kk += 32, buf ^= 1) {
    const bool hasNext = (kk + 32) < NFEAT;

    // ---- issue next tile's global loads (overlap with WMMA below) ----
    if (hasNext) {
      const float* srcA = xa + (size_t)(rowBase + tid) * NFEAT + kk + 32;
#pragma unroll
      for (int i = 0; i < 8; ++i) ar[i] = *(const float4*)(srcA + 4 * i);
#pragma unroll
      for (int i = 0; i < 16; ++i) {
        const int idx = tid + 128 * i;
        br[i] = W[(size_t)(kk + 32 + (idx >> 6)) * NFEAT + colBase + (idx & 63)];
      }
    }

    // ---- fragments from LDS (contiguous k = 16*half + s) + 8 WMMAs ----
    FragBF af[2];
#pragma unroll
    for (int p = 0; p < 2; ++p) {
      const unsigned int* Arow =
          (const unsigned int*)(As[buf] + (wave * 32 + p * 16 + m) * 32);
#pragma unroll
      for (int q = 0; q < 4; ++q) {
        af[p].u[q]     = Arow[half * 8 + q];      // K = half*16 + 2q
        af[p].u[q + 4] = Arow[half * 8 + 4 + q];  // K = half*16 + 8 + 2q
      }
    }
    FragBF bfr[4];
#pragma unroll
    for (int j = 0; j < 4; ++j) {
      const unsigned int* Brow = (const unsigned int*)(Bs[buf] + (j * 16 + m) * 32);
#pragma unroll
      for (int q = 0; q < 8; ++q) bfr[j].u[q] = Brow[half * 8 + q];  // K = half*16 + 2q
    }
#pragma unroll
    for (int p = 0; p < 2; ++p)
#pragma unroll
      for (int j = 0; j < 4; ++j)
        acc[p][j] = __builtin_amdgcn_wmma_f32_16x16x32_bf16(
            false, af[p].v, false, bfr[j].v, (short)0, acc[p][j], false, false);

    // ---- convert + store next tile into the other buffer ----
    if (hasNext) {
      unsigned short* dA = As[buf ^ 1] + tid * 32;
#pragma unroll
      for (int i = 0; i < 8; ++i) {
        dA[4 * i + 0] = f32_to_bf16(ar[i].x);
        dA[4 * i + 1] = f32_to_bf16(ar[i].y);
        dA[4 * i + 2] = f32_to_bf16(ar[i].z);
        dA[4 * i + 3] = f32_to_bf16(ar[i].w);
      }
#pragma unroll
      for (int i = 0; i < 16; ++i) {
        const int idx = tid + 128 * i;
        Bs[buf ^ 1][(idx & 63) * 32 + (idx >> 6)] = f32_to_bf16(br[i]);
      }
    }
    __syncthreads();
  }

  // ---- epilogue: C/D layout -> row = base + vgpr (+8 for hi lanes), col = lane ----
#pragma unroll
  for (int p = 0; p < 2; ++p) {
    const int r0 = rowBase + wave * 32 + p * 16 + half * 8;
#pragma unroll
    for (int j = 0; j < 4; ++j) {
      const int col = colBase + j * 16 + m;
      const float b = bias[col];
#pragma unroll
      for (int r = 0; r < 8; ++r)
        ya[(size_t)(r0 + r) * NFEAT + col] = acc[p][j][r] + b;
    }
  }
}

// ---------------------------------------------------------------------------
// Kernel 2: z_k = sum_t S[t]*w[f,P[t]] * gelu(x)[A[t]] * y[B[t]]; grade RMS norm
// one block per row n; 256 threads x 2 features
// ---------------------------------------------------------------------------
__global__ __launch_bounds__(256)
void ga_gp_norm_kernel(const float* __restrict__ x,
                       const float* __restrict__ gpw,
                       const float* __restrict__ y,
                       float* __restrict__ out) {
  const int n   = blockIdx.x;
  const int tid = threadIdx.x;
  __shared__ float red[4 * 256];

  float z[2][8];
  float gsum[4] = {0.f, 0.f, 0.f, 0.f};

#pragma unroll
  for (int ff = 0; ff < 2; ++ff) {
    const int f = tid + ff * 256;
    float xv[8], yv[8], gx[8];
#pragma unroll
    for (int a = 0; a < 8; ++a) {
      const size_t o = ((size_t)a * NBATCH + n) * NFEAT + f;
      xv[a] = x[o];
      yv[a] = y[o];
    }
#pragma unroll
    for (int a = 0; a < 8; ++a) {
      const float v = xv[a];
      gx[a] = 0.5f * v * (1.0f + erff(v * 0.70710678118654752440f));  // exact GELU
    }
    float w[20];
#pragma unroll
    for (int p = 0; p < 20; ++p) w[p] = gpw[f * 20 + p];
#pragma unroll
    for (int k = 0; k < 8; ++k) z[ff][k] = 0.f;
#pragma unroll
    for (int t = 0; t < 64; ++t)
      z[ff][TK[t]] = fmaf(TS[t] * w[TP[t]], gx[TA[t]] * yv[TB[t]], z[ff][TK[t]]);

    gsum[0] += z[ff][0] * z[ff][0];
    gsum[1] += z[ff][1] * z[ff][1] + z[ff][2] * z[ff][2] + z[ff][3] * z[ff][3];
    gsum[2] += z[ff][4] * z[ff][4] + z[ff][5] * z[ff][5] + z[ff][6] * z[ff][6];
    gsum[3] += z[ff][7] * z[ff][7];
  }

#pragma unroll
  for (int g = 0; g < 4; ++g) red[g * 256 + tid] = gsum[g];
  __syncthreads();
  for (int s = 128; s > 0; s >>= 1) {
    if (tid < s) {
#pragma unroll
      for (int g = 0; g < 4; ++g) red[g * 256 + tid] += red[g * 256 + tid + s];
    }
    __syncthreads();
  }
  float inv[4];
#pragma unroll
  for (int g = 0; g < 4; ++g)
    inv[g] = 1.0f / sqrtf(red[g * 256] * (1.0f / (float)NFEAT) + 1e-6f);

#pragma unroll
  for (int ff = 0; ff < 2; ++ff) {
    const int f = tid + ff * 256;
#pragma unroll
    for (int k = 0; k < 8; ++k)
      out[((size_t)k * NBATCH + n) * NFEAT + f] = z[ff][k] * inv[GRD[k]];
  }
}

// ---------------------------------------------------------------------------
extern "C" void kernel_launch(void* const* d_in, const int* in_sizes, int n_in,
                              void* d_out, int out_size, void* d_ws, size_t ws_size,
                              hipStream_t stream) {
  (void)in_sizes; (void)n_in; (void)out_size; (void)ws_size;
  const float* x    = (const float*)d_in[0];   // (8, 4096, 512)
  const float* lw   = (const float*)d_in[1];   // (4, 512, 512)
  const float* bias = (const float*)d_in[2];   // (512)
  const float* gpw  = (const float*)d_in[3];   // (512, 20)
  float*       out  = (float*)d_out;           // (8, 4096, 512)
  float*       yws  = (float*)d_ws;            // y intermediate: 64 MB

  dim3 g1(NFEAT / 64, NBATCH / 128, 8);
  ga_gemm_kernel<<<g1, 128, 0, stream>>>(x, lw, bias, yws);
  ga_gp_norm_kernel<<<NBATCH, 256, 0, stream>>>(x, gpw, yws, out);
}
